// PatchFeaturePropagation_15857019257406
// MI455X (gfx1250) — matrix operand order repaired and checked by reading, not compiled, not run on hardware
//
#include <hip/hip_runtime.h>
#include <hip/hip_bf16.h>

// ---------------------------------------------------------------------------
// PointNet++ FeaturePropagation on MI455X (gfx1250), bf16 WMMA pipeline.
// W-tile staging now uses the CDNA5 async global->LDS pipe (ASYNCcnt).
// ---------------------------------------------------------------------------

typedef __attribute__((ext_vector_type(16))) __bf16 v16bf;
typedef __attribute__((ext_vector_type(8)))  float  v8f;

static constexpr int Bb   = 16;
static constexpr int Nn   = 4096;
static constexpr int Ss   = 1024;
static constexpr int D1   = 256;
static constexpr int D2   = 512;
static constexpr int CIN  = D1 + D2;   // 768
static constexpr int C1   = 512;
static constexpr int C2   = 256;
static constexpr int Mm   = Bb * Nn;   // 65536

__device__ __forceinline__ unsigned short f32_to_bf16(float f) {
  unsigned int u = __float_as_uint(f);
  u += 0x7FFFu + ((u >> 16) & 1u);     // round-to-nearest-even
  return (unsigned short)(u >> 16);
}
__device__ __forceinline__ float bf16_to_f32(unsigned short h) {
  return __uint_as_float(((unsigned int)h) << 16);
}

union FragBF {        // one WMMA bf16 operand: 16 x bf16 = 8 VGPRs
  uint4 q[2];
  v16bf v;
};

// ---------------------------------------------------------------------------
// Prep kernels
// ---------------------------------------------------------------------------
__global__ void cvt_f32_bf16_kernel(const float* __restrict__ in,
                                    unsigned short* __restrict__ out, int n) {
  int i = blockIdx.x * 256 + threadIdx.x;
  if (i < n) out[i] = f32_to_bf16(in[i]);
}

// scale = g * rsqrt(v+eps); bias = (b - m) * scale + beta
__global__ void fold_bn_kernel(const float* __restrict__ b, const float* __restrict__ g,
                               const float* __restrict__ be, const float* __restrict__ m,
                               const float* __restrict__ v,
                               float* __restrict__ scale, float* __restrict__ bias, int n) {
  int i = blockIdx.x * 256 + threadIdx.x;
  if (i < n) {
    float s = g[i] * rsqrtf(v[i] + 1e-5f);
    scale[i] = s;
    bias[i]  = (b[i] - m[i]) * s + be[i];
  }
}

// ---------------------------------------------------------------------------
// 3-NN inverse-distance interpolation + concat -> Xcat bf16 [M, 768]
// grid: (Nn/256, Bb), block 256
// ---------------------------------------------------------------------------
__global__ __launch_bounds__(256) void interp_concat_kernel(
    const float* __restrict__ xyz1, const float* __restrict__ xyz2,
    const float* __restrict__ points1, const float* __restrict__ points2,
    unsigned short* __restrict__ xcat) {
  __shared__ float s_xyz2[Ss * 3];     // 12 KB
  __shared__ int   s_idx[256][3];
  __shared__ float s_wgt[256][3];

  const int t = threadIdx.x;
  const int b = blockIdx.y;
  const int nbase = blockIdx.x * 256;
  const int n = nbase + t;

  const float* x2 = xyz2 + (size_t)b * Ss * 3;
  for (int i = t; i < Ss * 3; i += 256) s_xyz2[i] = x2[i];
  __syncthreads();

  const float* p = xyz1 + ((size_t)b * Nn + n) * 3;
  const float px = p[0], py = p[1], pz = p[2];

  float d0 = 3.4e38f, d1 = 3.4e38f, d2 = 3.4e38f;
  int   i0 = 0, i1 = 0, i2 = 0;
  for (int s = 0; s < Ss; ++s) {
    float dx = px - s_xyz2[3 * s + 0];
    float dy = py - s_xyz2[3 * s + 1];
    float dz = pz - s_xyz2[3 * s + 2];
    float d = dx * dx + dy * dy + dz * dz;
    if (d < d2) {
      if (d < d1) {
        d2 = d1; i2 = i1;
        if (d < d0) { d1 = d0; i1 = i0; d0 = d; i0 = s; }
        else        { d1 = d;  i1 = s; }
      } else { d2 = d; i2 = s; }
    }
  }
  float r0 = 1.0f / (d0 + 1e-8f);
  float r1 = 1.0f / (d1 + 1e-8f);
  float r2 = 1.0f / (d2 + 1e-8f);
  float rs = 1.0f / (r0 + r1 + r2);
  s_idx[t][0] = i0; s_idx[t][1] = i1; s_idx[t][2] = i2;
  s_wgt[t][0] = r0 * rs; s_wgt[t][1] = r1 * rs; s_wgt[t][2] = r2 * rs;
  __syncthreads();

  const float* p2 = points2 + (size_t)b * Ss * D2;
  const float* p1 = points1 + (size_t)b * Nn * D1;

  // cooperative, coalesced gather/blend: one point at a time, all 256 lanes
  for (int pp = 0; pp < 256; ++pp) {
    const int   j0 = s_idx[pp][0], j1 = s_idx[pp][1], j2 = s_idx[pp][2];
    const float w0 = s_wgt[pp][0], w1 = s_wgt[pp][1], w2 = s_wgt[pp][2];
    const size_t row = ((size_t)b * Nn + nbase + pp) * CIN;
    const float* r0p = p2 + (size_t)j0 * D2;
    const float* r1p = p2 + (size_t)j1 * D2;
    const float* r2p = p2 + (size_t)j2 * D2;
    #pragma unroll
    for (int it = 0; it < D2 / 256; ++it) {
      int c = t + it * 256;
      float val = w0 * r0p[c] + w1 * r1p[c] + w2 * r2p[c];
      xcat[row + D1 + c] = f32_to_bf16(val);
    }
    // copy points1 channels (D1 == 256 == blockDim)
    xcat[row + t] = f32_to_bf16(p1[(size_t)(nbase + pp) * D1 + t]);
  }
}

// ---------------------------------------------------------------------------
// WMMA bf16 GEMM: out[m,o] = relu(scale[o]*(sum_k A[m,k]*W[o,k]) + bias[o])
// FINAL: also add residual (== A) and store f32.
// block = 256 threads = 8 waves, tile 128(M) x 64(N), K step 32.
// W tile staged via GLOBAL_LOAD_ASYNC_TO_LDS_B128 (ASYNCcnt pipe).
// ---------------------------------------------------------------------------
template <bool FINAL>
__global__ __launch_bounds__(256) void wmma_gemm_bn_relu(
    const unsigned short* __restrict__ A,    // [M,K] bf16
    const unsigned short* __restrict__ W,    // [O,K] bf16
    const float* __restrict__ scale, const float* __restrict__ bias,
    unsigned short* __restrict__ Ybf,        // [M,O] bf16   (non-final)
    float* __restrict__ Yf32,                // [M,O] f32    (final)
    const unsigned short* __restrict__ Res,  // [M,O] bf16   (final, == A)
    int M, int K, int O) {
  __shared__ uint4 ldsW[64][5];  // 64 rows x 32 bf16, padded to 80B rows

  const int t      = threadIdx.x;
  const int wave   = t >> 5;
  const int lane   = t & 31;
  const int laneHi = lane >> 4;      // 0: lanes 0-15, 1: lanes 16-31
  const int lane16 = lane & 15;

  const int m0 = blockIdx.x * 128 + wave * 16;
  const int n0 = blockIdx.y * 64;

  v8f acc[4] = {};                   // 4 x (16x16 f32)

  const int arow = m0 + lane16;      // A-matrix: lane row (both halves)
  const uint4* Aq = reinterpret_cast<const uint4*>(A) + (size_t)arow * (K / 8);

  const int wrow = t >> 2;           // cooperative W-tile stage: 1 x b128/lane
  const int wch  = t & 3;
  // per-lane LDS byte address of this thread's staging slot (low 32 bits of
  // the generic address of a __shared__ object are the LDS offset)
  const unsigned ldsOffW =
      (unsigned)(unsigned long long)(&ldsW[wrow][wch]);
  // per-lane global byte offset base within W for this thread's slot
  const unsigned wRowByte = (unsigned)((n0 + wrow) * K * 2 + wch * 16);

  for (int kb = 0; kb < K; kb += 32) {
    __syncthreads();   // previous iteration's LDS reads done before restage
    {
      // async DMA: W[(n0+wrow), kb + wch*8 .. +8) -> ldsW[wrow][wch]
      const unsigned gofs = wRowByte + (unsigned)(kb * 2);
      asm volatile("global_load_async_to_lds_b128 %0, %1, %2"
                   :
                   : "v"(ldsOffW), "v"(gofs), "s"(W)
                   : "memory");
      asm volatile("s_wait_asynccnt 0" ::: "memory");
    }
    __syncthreads();   // all waves' async stages visible in LDS

    // A fragment: lane<16 holds K {0..7,16..23}, lane>=16 holds {8..15,24..31}
    FragBF a;
    const int abase = kb / 8 + laneHi;
    a.q[0] = Aq[abase];
    a.q[1] = Aq[abase + 2];
    if (kb + 32 < K)
      __builtin_prefetch((const void*)(Aq + abase + 4), 0, 1);  // global_prefetch

    #pragma unroll
    for (int ng = 0; ng < 4; ++ng) {
      // B fragment: lane = output column; lane<16 K {0..7,16..23}, lane>=16 {8..15,24..31}
      FragBF bw;
      const int brow = ng * 16 + lane16;
      bw.q[0] = ldsW[brow][laneHi];
      bw.q[1] = ldsW[brow][laneHi + 2];
      acc[ng] = __builtin_amdgcn_wmma_f32_16x16x32_bf16(
          false, a.v, false, bw.v, (short)0, acc[ng], false, false);
    }
  }

  // Epilogue per C layout: VGPR r, lanes0-15 -> (M=r, N=lane); 16-31 -> (M=r+8)
  #pragma unroll
  for (int ng = 0; ng < 4; ++ng) {
    const int col = n0 + ng * 16 + lane16;
    const float sc = scale[col];
    const float bi = bias[col];
    #pragma unroll
    for (int r = 0; r < 8; ++r) {
      const int row = m0 + r + laneHi * 8;
      const size_t off = (size_t)row * O + col;
      float h = fmaxf(acc[ng][r] * sc + bi, 0.0f);
      if (FINAL) {
        float x = bf16_to_f32(Res[off]);
        Yf32[off] = fmaxf(h + x, 0.0f);
      } else {
        Ybf[off] = f32_to_bf16(h);
      }
    }
  }
}

// ---------------------------------------------------------------------------
// Host launcher
// ---------------------------------------------------------------------------
extern "C" void kernel_launch(void* const* d_in, const int* in_sizes, int n_in,
                              void* d_out, int out_size, void* d_ws, size_t ws_size,
                              hipStream_t stream) {
  (void)in_sizes; (void)n_in; (void)out_size; (void)ws_size;

  const float* xyz1    = (const float*)d_in[0];
  const float* xyz2    = (const float*)d_in[1];
  const float* points1 = (const float*)d_in[2];
  const float* points2 = (const float*)d_in[3];
  const float* w1 = (const float*)d_in[4];
  const float* b1 = (const float*)d_in[5];
  const float* g1 = (const float*)d_in[6];
  const float* be1 = (const float*)d_in[7];
  const float* m1 = (const float*)d_in[8];
  const float* v1 = (const float*)d_in[9];
  const float* w2 = (const float*)d_in[10];
  const float* b2 = (const float*)d_in[11];
  const float* g2 = (const float*)d_in[12];
  const float* be2 = (const float*)d_in[13];
  const float* m2 = (const float*)d_in[14];
  const float* v2 = (const float*)d_in[15];
  const float* wr = (const float*)d_in[16];
  const float* br = (const float*)d_in[17];
  const float* gr = (const float*)d_in[18];
  const float* ber = (const float*)d_in[19];
  const float* mr = (const float*)d_in[20];
  const float* vr = (const float*)d_in[21];

  char* ws = (char*)d_ws;
  size_t off = 0;
  auto take = [&](size_t bytes) -> void* {
    void* p = (void*)(ws + off);
    off += (bytes + 255) & ~(size_t)255;
    return p;
  };

  unsigned short* xcat = (unsigned short*)take((size_t)Mm * CIN * 2);  // 96 MB
  unsigned short* y1   = (unsigned short*)take((size_t)Mm * C1 * 2);   // 64 MB
  unsigned short* y2   = (unsigned short*)take((size_t)Mm * C2 * 2);   // 32 MB
  unsigned short* w1b  = (unsigned short*)take((size_t)C1 * CIN * 2);
  unsigned short* w2b  = (unsigned short*)take((size_t)C2 * C1 * 2);
  unsigned short* wrb  = (unsigned short*)take((size_t)C2 * C2 * 2);
  float* s1 = (float*)take(C1 * 4); float* t1 = (float*)take(C1 * 4);
  float* s2 = (float*)take(C2 * 4); float* t2 = (float*)take(C2 * 4);
  float* sr = (float*)take(C2 * 4); float* tr = (float*)take(C2 * 4);

  // weight conversion + BN folding
  cvt_f32_bf16_kernel<<<(C1 * CIN + 255) / 256, 256, 0, stream>>>(w1, w1b, C1 * CIN);
  cvt_f32_bf16_kernel<<<(C2 * C1 + 255) / 256, 256, 0, stream>>>(w2, w2b, C2 * C1);
  cvt_f32_bf16_kernel<<<(C2 * C2 + 255) / 256, 256, 0, stream>>>(wr, wrb, C2 * C2);
  fold_bn_kernel<<<(C1 + 255) / 256, 256, 0, stream>>>(b1, g1, be1, m1, v1, s1, t1, C1);
  fold_bn_kernel<<<(C2 + 255) / 256, 256, 0, stream>>>(b2, g2, be2, m2, v2, s2, t2, C2);
  fold_bn_kernel<<<(C2 + 255) / 256, 256, 0, stream>>>(br, gr, ber, mr, vr, sr, tr, C2);

  // 3-NN interpolation + concat -> bf16 X [M, 768]
  interp_concat_kernel<<<dim3(Nn / 256, Bb), 256, 0, stream>>>(
      xyz1, xyz2, points1, points2, xcat);

  // GEMM1: [M,768] x [512,768]^T -> y1 bf16
  wmma_gemm_bn_relu<false><<<dim3(Mm / 128, C1 / 64), 256, 0, stream>>>(
      xcat, w1b, s1, t1, y1, nullptr, nullptr, Mm, CIN, C1);
  // GEMM2: [M,512] x [256,512]^T -> y2 bf16
  wmma_gemm_bn_relu<false><<<dim3(Mm / 128, C2 / 64), 256, 0, stream>>>(
      y1, w2b, s2, t2, y2, nullptr, nullptr, Mm, C1, C2);
  // GEMM3 (+BN+ReLU+residual+ReLU) -> f32 out
  wmma_gemm_bn_relu<true><<<dim3(Mm / 128, C2 / 64), 256, 0, stream>>>(
      y2, wrb, sr, tr, nullptr, (float*)d_out, y2, Mm, C2, C2);
}
